// HistogramLoss_21062519620065
// MI455X (gfx1250) — compile-verified, matrix operand order checked by reading, not run on hardware
//
#include <hip/hip_runtime.h>
#include <hip/hip_bf16.h>

// ---------------------------------------------------------------------------
// HistogramLoss for MI455X (gfx1250, wave32, WMMA)
//
//  F = 256 features, P = 64*64 = 4096 pixels, BINS = 51, classes 1..18.
//
//  Kernel 1: masked moment GEMMs (miu_raw, m2_raw) via v_wmma_f32_16x16x32_f16
//  Kernel 2: per-class KDE histogram + smooth-L1 (exp-bound, TRANS units)
//  Kernel 3: deterministic final reduction over 18 classes
// ---------------------------------------------------------------------------

#define NUM_F     256
#define NUM_P     4096
#define BINS      51
#define NUM_CLS   19      // stats computed for 0..18; loss uses 1..18

typedef __attribute__((ext_vector_type(16))) _Float16 v16h;
typedef __attribute__((ext_vector_type(8)))  float    v8f;

// ---------------------------------------------------------------------------
// Kernel 1: miu_raw[c][f] = sum_p x[f,p] * (lab[p]==c)
//           m2_raw [c][f] = sum_p x[f,p]^2 * (lab[p]==c)
// One wave (32 lanes) per 16x16 output tile. grid = (16 f-tiles, 2 c-tiles).
// A: 16(f) x 32(p) f16, B: 32(p) x 16(c) 0/1 f16, C/D: f32 accum.
//
// wave32 16-bit A layout (ISA 7.12.2): lane L holds row M = L&15;
//   halfs 0..7  -> K = kb + 0..7,  halfs 8..15 -> K = kb + 16..23,
//   kb = 0 for lanes 0..15, kb = 8 for lanes 16..31.
// B mirrors with N in place of M. D (f32): VGPR r, lanes 0..15 -> M=r,
//   lanes 16..31 -> M=r+8, N = lane&15.
// ---------------------------------------------------------------------------
__global__ __launch_bounds__(32)
void hist_stats_wmma(const float* __restrict__ x, const int* __restrict__ lab,
                     float* __restrict__ miu_raw, float* __restrict__ m2_raw) {
    const int ftile = blockIdx.x;          // 0..15
    const int ctile = blockIdx.y;          // 0..1
    const int lane  = threadIdx.x;         // 0..31
    const int rowm  = lane & 15;
    const int kb    = (lane < 16) ? 0 : 8;
    const int f     = ftile * 16 + rowm;   // A-matrix row for this lane
    const int ccol  = ctile * 16 + rowm;   // B-matrix column for this lane

    v8f acc_miu = {};
    v8f acc_m2  = {};

    for (int k0 = 0; k0 < NUM_P; k0 += 32) {
        // stream hint for the next chunk of this lane's row
        __builtin_prefetch(x + f * NUM_P + k0 + 32, 0, 3);

        v16h a, a2, b;
#pragma unroll
        for (int i = 0; i < 8; ++i) {
            const int k_lo = k0 + kb + i;
            const int k_hi = k0 + kb + 16 + i;
            const float xl = x[f * NUM_P + k_lo];
            const float xh = x[f * NUM_P + k_hi];
            a [i]     = (_Float16)xl;
            a [i + 8] = (_Float16)xh;
            a2[i]     = (_Float16)(xl * xl);
            a2[i + 8] = (_Float16)(xh * xh);
            b [i]     = (lab[k_lo] == ccol) ? (_Float16)1.0f : (_Float16)0.0f;
            b [i + 8] = (lab[k_hi] == ccol) ? (_Float16)1.0f : (_Float16)0.0f;
        }
        acc_miu = __builtin_amdgcn_wmma_f32_16x16x32_f16(
            false, a,  false, b, (short)0, acc_miu, false, false);
        acc_m2  = __builtin_amdgcn_wmma_f32_16x16x32_f16(
            false, a2, false, b, (short)0, acc_m2,  false, false);
    }

    const int c_out = ctile * 16 + (lane & 15);
    const int mofs  = (lane >= 16) ? 8 : 0;
    if (c_out < NUM_CLS) {
#pragma unroll
        for (int r = 0; r < 8; ++r) {
            const int f_out = ftile * 16 + r + mofs;
            miu_raw[c_out * NUM_F + f_out] = acc_miu[r];
            m2_raw [c_out * NUM_F + f_out] = acc_m2[r];
        }
    }
}

// ---------------------------------------------------------------------------
// Kernel 2: per-class KDE histogram + smooth-L1. One block per class c=1..18,
// 256 threads = one feature per thread. 51 KDE accumulators fully unrolled so
// they live in VGPRs; exp() on the TRANS pipes co-executes with the VALU adds.
// ---------------------------------------------------------------------------
__global__ __launch_bounds__(256)
void hist_class_loss(const float* __restrict__ x, const int* __restrict__ lab,
                     const float* __restrict__ miu_raw,
                     const float* __restrict__ m2_raw,
                     float* __restrict__ cls_loss, float* __restrict__ cls_act) {
    const int c = blockIdx.x + 1;          // classes 1..18
    const int t = threadIdx.x;             // feature index

    __shared__ int   s_lab[NUM_P];         // 16 KB of 320 KB LDS
    __shared__ float s_red[256];
    __shared__ float s_n;

    for (int i = t; i < NUM_P; i += 256) s_lab[i] = lab[i];
    __syncthreads();

    // n = count of pixels in class c (deterministic tree reduce)
    int cnt = 0;
    for (int i = t; i < NUM_P; i += 256) cnt += (s_lab[i] == c) ? 1 : 0;
    s_red[t] = (float)cnt;
    __syncthreads();
    for (int s = 128; s > 0; s >>= 1) {
        if (t < s) s_red[t] += s_red[t + s];
        __syncthreads();
    }
    if (t == 0) s_n = s_red[0];
    __syncthreads();

    const float n      = s_n;
    const float n_safe = fmaxf(n, 1.0f);

    // Per-feature Gaussian stats (stop-gradient in the reference)
    const float miu   = miu_raw[c * NUM_F + t] / n_safe;
    const float m2    = m2_raw [c * NUM_F + t] / n_safe;
    const float var   = fmaxf(m2 - miu * miu, 1e-12f);
    const float var_s = var * (1.0f / 25.0f);      // KDE bandwidth^2
    const float TWO_PI = 6.28318530717958647692f;

    // ---- KDE accumulation over this class's pixels (dominant cost) ----
    float acc[BINS];
#pragma unroll
    for (int b = 0; b < BINS; ++b) acc[b] = 0.0f;

    const float ninv2vs = -0.5f / var_s;
    const float* __restrict__ xrow = x + (size_t)t * NUM_P;

    for (int p = 0; p < NUM_P; p += 32) {
        // keep this lane's contiguous stream one 128B line ahead
        // (global_prefetch_b8: no LOADcnt, no VGPR return)
        __builtin_prefetch(xrow + p + 32, 0, 3);
#pragma unroll
        for (int q = 0; q < 32; q += 4) {
            const float4 xv = *(const float4*)(xrow + p + q);
            const float xs[4] = {xv.x, xv.y, xv.z, xv.w};
#pragma unroll
            for (int j = 0; j < 4; ++j) {
                if (s_lab[p + q + j] == c) {       // uniform across the block
                    const float xx = xs[j];
#pragma unroll
                    for (int b = 0; b < BINS; ++b) {
                        const float bin = -5.0f + 0.2f * (float)b;
                        const float d   = bin - xx;
                        acc[b] += __expf(d * d * ninv2vs);  // v_exp_f32 (TRANS)
                    }
                }
            }
        }
    }

    // hist = sv / max(sum(sv), 1e-12), sv = acc / sqrt(2*pi*var_s)
    float ssum = 0.0f;
#pragma unroll
    for (int b = 0; b < BINS; ++b) ssum += acc[b];
    const float inv_sqrt_vs = rsqrtf(TWO_PI * var_s);
    const float hist_den    = fmaxf(ssum * inv_sqrt_vs, 1e-12f);

    // target: Gaussian pdf at bins normalized over bins (1/sqrt(2*pi*var)
    // cancels in the normalization)
    const float ninv2v = -0.5f / var;
    float tsum = 0.0f;
#pragma unroll
    for (int b = 0; b < BINS; ++b) {
        const float d = (-5.0f + 0.2f * (float)b) - miu;
        tsum += __expf(d * d * ninv2v);
    }
    const float tden = fmaxf(tsum, 1e-30f);

    // smooth-L1 (beta = 1), per-thread partial over 51 bins
    float lsum = 0.0f;
#pragma unroll
    for (int b = 0; b < BINS; ++b) {
        const float d      = (-5.0f + 0.2f * (float)b) - miu;
        const float target = __expf(d * d * ninv2v) / tden;
        const float hist   = (acc[b] * inv_sqrt_vs) / hist_den;
        const float diff   = hist - target;
        const float ad     = fabsf(diff);
        lsum += (ad < 1.0f) ? 0.5f * diff * diff : (ad - 0.5f);
    }

    // deterministic block reduction over the 256 features
    s_red[t] = lsum;
    __syncthreads();
    for (int s = 128; s > 0; s >>= 1) {
        if (t < s) s_red[t] += s_red[t + s];
        __syncthreads();
    }
    if (t == 0) {
        const float has = (n > 0.0f) ? 1.0f : 0.0f;
        cls_loss[blockIdx.x] = (s_red[0] * (1.0f / (float)(NUM_F * BINS))) * has; // LOSS_WEIGHT=1
        cls_act [blockIdx.x] = has;
    }
}

// ---------------------------------------------------------------------------
// Kernel 3: fixed-order final reduction over the 18 classes
// ---------------------------------------------------------------------------
__global__ __launch_bounds__(32)
void hist_finalize(const float* __restrict__ cls_loss,
                   const float* __restrict__ cls_act,
                   float* __restrict__ out) {
    if (threadIdx.x == 0 && blockIdx.x == 0) {
        float l = 0.0f, a = 0.0f;
        for (int i = 0; i < 18; ++i) { l += cls_loss[i]; a += cls_act[i]; }
        out[0] = l / (a + 1e-12f);
    }
}

// ---------------------------------------------------------------------------
extern "C" void kernel_launch(void* const* d_in, const int* in_sizes, int n_in,
                              void* d_out, int out_size, void* d_ws, size_t ws_size,
                              hipStream_t stream) {
    const float* feat = (const float*)d_in[0];   // [1,256,64,64] f32
    const int*   lab  = (const int*)  d_in[1];   // [1,1,64,64]

    float* ws       = (float*)d_ws;
    float* cls_loss = ws;                        // [18]
    float* cls_act  = ws + 18;                   // [18]
    float* miu_raw  = ws + 64;                   // [19][256]
    float* m2_raw   = miu_raw + NUM_CLS * NUM_F; // [19][256]

    dim3 gA(16, 2);
    hist_stats_wmma<<<gA, 32, 0, stream>>>(feat, lab, miu_raw, m2_raw);
    hist_class_loss<<<18, 256, 0, stream>>>(feat, lab, miu_raw, m2_raw,
                                            cls_loss, cls_act);
    hist_finalize<<<1, 32, 0, stream>>>(cls_loss, cls_act, (float*)d_out);
}
